// MaskedAttention_69544110456945
// MI455X (gfx1250) — compile-verified, matrix operand order checked by reading, not run on hardware
//
#include <hip/hip_runtime.h>
#include <math.h>
#include <stdint.h>

typedef _Float16 h16;
typedef __attribute__((ext_vector_type(16))) _Float16 v16h;
typedef __attribute__((ext_vector_type(8)))  _Float16 v8h;
typedef __attribute__((ext_vector_type(8)))  float    v8f;

#define D_MODEL 1024
#define D_HEAD  128
#define BATCH   4
#define SEQ     4096
#define BS      (BATCH*SEQ)

// Build a 16x16x32 f16 A/B operand (16 halves per lane) from a base pointer
// that already points at this lane's row/col start + kb0 (0 or 8).
// Halves [0..7] are K = kb0..kb0+7, halves [8..15] are K = kb0+16..kb0+23,
// matching the CDNA5 16-bit 16x32 operand layout.
__device__ __forceinline__ v16h make_op(const h16* base) {
  v8h lo = *(const v8h*)(base);
  v8h hi = *(const v8h*)(base + 16);
  v16h r;
#pragma unroll
  for (int j = 0; j < 8; ++j) { r[j] = lo[j]; r[j + 8] = hi[j]; }
  return r;
}

// ---------------------------------------------------------------------------
// Kernel 1: fused QKV projection.  x[16384,1024] fp32 -> q/k/v [16384,128] f16.
// Block = 256 thr (8 waves), 64 rows per block.  Wave w owns col tile w (16
// cols) of each of q,k,v across 4 row tiles -> 12 f32 accumulators.
// K loop: stage x tile (64x32, f16) + W slice (32x384, f16, col-major) in LDS.
// ---------------------------------------------------------------------------
__global__ __launch_bounds__(256)
void qkv_proj_kernel(const float* __restrict__ x,
                     const float* __restrict__ Wq, const float* __restrict__ bq,
                     const float* __restrict__ Wk, const float* __restrict__ bk,
                     const float* __restrict__ Wv, const float* __restrict__ bv,
                     h16* __restrict__ qf, h16* __restrict__ kf, h16* __restrict__ vf)
{
  __shared__ h16 As[64 * 40];    // x tile, row-major, stride 40 halves (80B)
  __shared__ h16 Ws[384 * 40];   // W slice, [col][k], stride 40 halves

  const int t    = threadIdx.x;
  const int w    = t >> 5;       // wave id 0..7
  const int lane = t & 31;
  const int half = lane >> 4;
  const int ln   = lane & 15;
  const int kb0  = half ? 8 : 0;
  const int row0 = blockIdx.x * 64;

  v8f vzero = {};
  v8f acc[3][4];
#pragma unroll
  for (int m = 0; m < 3; ++m)
#pragma unroll
    for (int rt = 0; rt < 4; ++rt) acc[m][rt] = vzero;

  const float* Wm[3] = {Wq, Wk, Wv};

  for (int kc = 0; kc < D_MODEL; kc += 32) {
    __syncthreads();
    // stage x tile: 64x32 fp32 -> f16 (2048 elems, 8 per thread, coalesced)
#pragma unroll
    for (int i = 0; i < 8; ++i) {
      int idx = t + 256 * i;
      int r = idx >> 5, k = idx & 31;
      As[r * 40 + k] = (h16)x[(size_t)(row0 + r) * D_MODEL + kc + k];
    }
    // stage W slice: 3 x (32x128) fp32 -> f16 transposed to [col][k]
#pragma unroll
    for (int m = 0; m < 3; ++m) {
      const float* Wp = Wm[m];
#pragma unroll
      for (int i = 0; i < 16; ++i) {
        int idx = t + 256 * i;
        int k = idx >> 7, c = idx & 127;
        Ws[(m * 128 + c) * 40 + k] = (h16)Wp[(size_t)(kc + k) * D_HEAD + c];
      }
    }
    __syncthreads();

    v16h bmat[3];
#pragma unroll
    for (int m = 0; m < 3; ++m)
      bmat[m] = make_op(&Ws[(m * 128 + w * 16 + ln) * 40 + kb0]);

#pragma unroll
    for (int rt = 0; rt < 4; ++rt) {
      v16h a = make_op(&As[(rt * 16 + ln) * 40 + kb0]);
#pragma unroll
      for (int m = 0; m < 3; ++m)
        acc[m][rt] = __builtin_amdgcn_wmma_f32_16x16x32_f16(
            false, a, false, bmat[m], (short)0, acc[m][rt], false, false);
    }
  }

  // epilogue: add bias, store f16 q/k/v
  const float* biasArr[3] = {bq, bk, bv};
  h16* outArr[3] = {qf, kf, vf};
  const int col = w * 16 + ln;
#pragma unroll
  for (int m = 0; m < 3; ++m) {
    float bias = biasArr[m][col];
    h16* op = outArr[m];
#pragma unroll
    for (int rt = 0; rt < 4; ++rt)
#pragma unroll
      for (int r = 0; r < 8; ++r) {
        int rowg = row0 + rt * 16 + r + 8 * half;
        op[(size_t)rowg * D_HEAD + col] = (h16)(acc[m][rt][r] + bias);
      }
  }
}

// ---------------------------------------------------------------------------
// Kernel 2: causal flash attention, software-pipelined.
// Block = 256 thr (8 waves), 128 query rows per block, 32 keys per step.
// K tiles: double-buffered in LDS, filled by GLOBAL_LOAD_ASYNC_TO_LDS_B128
//   (ASYNCcnt), prefetched one iteration ahead.
// V tiles: prefetched to VGPRs one iteration ahead, stored transposed to a
//   double-buffered LDS tile.
// One __syncthreads + one s_wait_asynccnt per iteration; the P bounce relies
// on same-wave DScnt ordering (no barrier).
// ---------------------------------------------------------------------------
__global__ __launch_bounds__(256)
void attn_kernel(const h16* __restrict__ qf, const h16* __restrict__ kf,
                 const h16* __restrict__ vf, float* __restrict__ out)
{
  __shared__ h16 Ks[2][32 * 136];   // K tile [key][d], stride 136 halves
  __shared__ h16 VsT[2][128 * 40];  // V tile transposed [d][key], stride 40
  __shared__ h16 Ps[8][16 * 40];    // per-wave P bounce [row][key], stride 40

  const int t    = threadIdx.x;
  const int wq   = t >> 5;
  const int lane = t & 31;
  const int half = lane >> 4;
  const int ln   = lane & 15;
  const int kb0  = half ? 8 : 0;
  const int qb   = blockIdx.x;
  const int b    = blockIdx.y;

  const size_t bofs  = (size_t)b * SEQ * D_HEAD;
  const int    qrow0 = qb * 128 + wq * 16;   // first seq position of this wave

  const h16* kbase = kf + bofs;
  const h16* vbase = vf + bofs;

  // cooperative staging coordinates: thread t owns (key row, 16-col chunk)
  const int krow = t >> 3;            // 0..31
  const int c0   = (t & 7) * 16;      // 0..112

  // Async copy of this thread's 32B of the K tile for block j into buffer bb.
  // IOFFSET applies to both the global and the LDS address, so two issues
  // with offset 0/16 cover the 16 contiguous halves.
  auto prefetchK = [&](int j, int bb) {
    const h16* kp = kbase + (size_t)(j * 32) * D_HEAD + krow * D_HEAD + c0;
    unsigned lds = (unsigned)(size_t)&Ks[bb][krow * 136 + c0];
    asm volatile("global_load_async_to_lds_b128 %0, %1, off"
                 :: "v"(lds), "v"(kp) : "memory");
    asm volatile("global_load_async_to_lds_b128 %0, %1, off offset:16"
                 :: "v"(lds), "v"(kp) : "memory");
  };
  auto loadV = [&](int j, v8h& b0, v8h& b1) {
    const h16* vp = vbase + (size_t)(j * 32) * D_HEAD + krow * D_HEAD + c0;
    b0 = *(const v8h*)(vp);
    b1 = *(const v8h*)(vp + 8);
  };
  auto storeVT = [&](int bb, v8h b0, v8h b1) {
#pragma unroll
    for (int j = 0; j < 8; ++j) {
      VsT[bb][(c0 + j)     * 40 + krow] = b0[j];
      VsT[bb][(c0 + 8 + j) * 40 + krow] = b1[j];
    }
  };

  // load Q operands (4 chunks of K=32 over d_head=128), pre-scaled by 1/sqrt(d)
  v16h qa[4];
  {
    const h16* qp = qf + bofs + (size_t)(qrow0 + ln) * D_HEAD;
    const h16 sc = (h16)0.08838834764831845f;   // 1/sqrt(128)
#pragma unroll
    for (int c = 0; c < 4; ++c) {
      v16h v = make_op(qp + c * 32 + kb0);
#pragma unroll
      for (int j = 0; j < 16; ++j) v[j] = (h16)(v[j] * sc);
      qa[c] = v;
    }
  }

  v8f vzero = {};
  v8f o[8];
#pragma unroll
  for (int c = 0; c < 8; ++c) o[c] = vzero;
  float mrow[8], lrow[8];
#pragma unroll
  for (int r = 0; r < 8; ++r) { mrow[r] = -1e30f; lrow[r] = 0.f; }

  const int nkb = qb * 4 + 4;   // causal: key blocks up to end of query block

  // ---- pipeline preamble: block 0 in flight, V(1) in registers ----
  v8h vr0, vr1;
  prefetchK(0, 0);
  loadV(0, vr0, vr1);
  storeVT(0, vr0, vr1);
  loadV(nkb > 1 ? 1 : 0, vr0, vr1);

  for (int kbi = 0; kbi < nkb; ++kbi) {
    const int buf = kbi & 1;
    const int nxt = (kbi + 1 < nkb) ? (kbi + 1) : kbi;

    // own async batch for block kbi is the only one outstanding
    asm volatile("s_wait_asynccnt 0x0" ::: "memory");
    __syncthreads();   // all waves' K(kbi) + V(kbi) tiles resident; buffers safe

    // prefetch block kbi+1 (clamped duplicate on the last iteration)
    prefetchK(nxt, buf ^ 1);
    storeVT(buf ^ 1, vr0, vr1);                 // V(kbi+1) -> LDS
    loadV((kbi + 2 < nkb) ? (kbi + 2) : nxt, vr0, vr1);  // V(kbi+2) -> regs

    // ---- scores: two 16x16 f32 tiles (keys 0..15 / 16..31 of the block) ----
    v8f s0 = vzero, s1 = vzero;
#pragma unroll
    for (int c = 0; c < 4; ++c) {
      v16h bk0 = make_op(&Ks[buf][ln * 136 + c * 32 + kb0]);
      v16h bk1 = make_op(&Ks[buf][(16 + ln) * 136 + c * 32 + kb0]);
      s0 = __builtin_amdgcn_wmma_f32_16x16x32_f16(false, qa[c], false, bk0,
                                                  (short)0, s0, false, false);
      s1 = __builtin_amdgcn_wmma_f32_16x16x32_f16(false, qa[c], false, bk1,
                                                  (short)0, s1, false, false);
    }

    // ---- causal mask + online softmax update ----
    const int kcol0 = kbi * 32;
#pragma unroll
    for (int r = 0; r < 8; ++r) {
      int qr = qrow0 + r + 8 * half;
      float v0 = (kcol0 + ln      <= qr) ? s0[r] : -1e30f;
      float v1 = (kcol0 + 16 + ln <= qr) ? s1[r] : -1e30f;
      float vmax = fmaxf(v0, v1);
#pragma unroll
      for (int off = 1; off < 16; off <<= 1)
        vmax = fmaxf(vmax, __shfl_xor(vmax, off, 16));
      float mnew  = fmaxf(mrow[r], vmax);
      float alpha = __expf(mrow[r] - mnew);
      float p0 = __expf(v0 - mnew);
      float p1 = __expf(v1 - mnew);
      float psum = p0 + p1;
#pragma unroll
      for (int off = 1; off < 16; off <<= 1)
        psum += __shfl_xor(psum, off, 16);
      lrow[r] = lrow[r] * alpha + psum;
      mrow[r] = mnew;
#pragma unroll
      for (int c = 0; c < 8; ++c) o[c][r] *= alpha;
      Ps[wq][(r + 8 * half) * 40 + ln]      = (h16)p0;
      Ps[wq][(r + 8 * half) * 40 + 16 + ln] = (h16)p1;
    }

    // ---- O += P V : per-wave P bounce ordered by DScnt (no barrier) ----
    v16h pa = make_op(&Ps[wq][ln * 40 + kb0]);
#pragma unroll
    for (int c = 0; c < 8; ++c) {
      v16h bvv = make_op(&VsT[buf][(c * 16 + ln) * 40 + kb0]);
      o[c] = __builtin_amdgcn_wmma_f32_16x16x32_f16(false, pa, false, bvv,
                                                    (short)0, o[c], false, false);
    }
  }

  // drain the last (dummy) async batch before the wave ends
  asm volatile("s_wait_asynccnt 0x0" ::: "memory");

  // ---- finalize: O / l, store fp32 ----
  float rl[8];
#pragma unroll
  for (int r = 0; r < 8; ++r) rl[r] = 1.0f / lrow[r];
  float* op = out + bofs + (size_t)qrow0 * D_HEAD;
#pragma unroll
  for (int c = 0; c < 8; ++c)
#pragma unroll
    for (int r = 0; r < 8; ++r)
      op[(size_t)(r + 8 * half) * D_HEAD + c * 16 + ln] = o[c][r] * rl[r];
}

extern "C" void kernel_launch(void* const* d_in, const int* in_sizes, int n_in,
                              void* d_out, int out_size, void* d_ws, size_t ws_size,
                              hipStream_t stream) {
  const float* x  = (const float*)d_in[0];
  const float* Wq = (const float*)d_in[1];
  const float* bq = (const float*)d_in[2];
  const float* Wk = (const float*)d_in[3];
  const float* bk = (const float*)d_in[4];
  const float* Wv = (const float*)d_in[5];
  const float* bv = (const float*)d_in[6];
  float* out = (float*)d_out;

  // f16 Q/K/V intermediates in workspace: 3 x 16384 x 128 halves = 12 MB
  h16* qf = (h16*)d_ws;
  h16* kf = qf + (size_t)BS * D_HEAD;
  h16* vf = kf + (size_t)BS * D_HEAD;

  qkv_proj_kernel<<<BS / 64, 256, 0, stream>>>(x, Wq, bq, Wk, bk, Wv, bv,
                                               qf, kf, vf);
  attn_kernel<<<dim3(SEQ / 128, BATCH), 256, 0, stream>>>(qf, kf, vf, out);
}
